// Attention_20667382628896
// MI455X (gfx1250) — matrix operand order repaired and checked by reading, not compiled, not run on hardware
//
#include <hip/hip_runtime.h>
#include <hip/hip_bf16.h>
#include <stdint.h>

#define HEADS   12
#define S_LEN   2048
#define DMODEL  768
#define DHEAD   64
#define BATCH   2

typedef __attribute__((ext_vector_type(16))) __bf16 v16bf;
typedef __attribute__((ext_vector_type(8)))  float  v8f;

union FragB {
    v16bf v;
    uint4 q[2];
    unsigned short h[16];
};

// fp32 -> bf16 via native conversion
__device__ __forceinline__ unsigned short f2bf(float f) {
    __bf16 b = (__bf16)f;
    return __builtin_bit_cast(unsigned short, b);
}

__device__ __forceinline__ v8f wmma_bf16(const FragB& a, const FragB& b, v8f c) {
    return __builtin_amdgcn_wmma_f32_16x16x32_bf16(
        /*neg_a=*/false, a.v, /*neg_b=*/false, b.v,
        /*c_mod=*/(short)0, c, /*reuse_a=*/false, /*reuse_b=*/false);
}

// VALU lane-xor within 16-lane rows via v_permlane16_b32 (no LDS round-trip)
__device__ __forceinline__ float pl16(float x, unsigned s1, unsigned s2) {
    unsigned xi = __builtin_bit_cast(unsigned, x);
    unsigned r  = __builtin_amdgcn_permlane16(xi, xi, s1, s2, false, false);
    return __builtin_bit_cast(float, r);
}
// butterfly max over the 16 lanes of each half (rows live per-half)
__device__ __forceinline__ float rowmax16(float x) {
    x = fmaxf(x, pl16(x, 0x67452301u, 0xEFCDAB89u));   // xor 1
    x = fmaxf(x, pl16(x, 0x54761032u, 0xDCFE98BAu));   // xor 2
    x = fmaxf(x, pl16(x, 0x32107654u, 0xBA98FEDCu));   // xor 4
    x = fmaxf(x, pl16(x, 0xFEDCBA98u, 0x76543210u));   // xor 8
    return x;
}
__device__ __forceinline__ float rowsum16(float x) {
    x += pl16(x, 0x67452301u, 0xEFCDAB89u);
    x += pl16(x, 0x54761032u, 0xDCFE98BAu);
    x += pl16(x, 0x32107654u, 0xBA98FEDCu);
    x += pl16(x, 0xFEDCBA98u, 0x76543210u);
    return x;
}

// ---------------------------------------------------------------------------
// Projection: C[M=4096, N=768] = X[4096,768] * W[768,768]  (fp32 in, bf16 out)
// VTRANS==0 : Out[b][h][s][d]   (for Q, K)
// VTRANS==1 : Out[b][h][d][s]   (for V, transposed so V feeds WMMA B operand)
// Block tile 128x128, 8 waves, each wave 32x64 (2x4 WMMA 16x16 frags).
// ---------------------------------------------------------------------------
template <int VTRANS>
__global__ __launch_bounds__(256) void proj_kernel(
    const float* __restrict__ X, const float* __restrict__ W,
    unsigned short* __restrict__ Out)
{
    __shared__ unsigned short lA[128 * 48];   // [m][k], k-stride 48 (96B, 16B aligned)
    __shared__ unsigned short lB[128 * 48];   // [n][k] (W tile transposed)

    const int tid  = threadIdx.x;
    const int lane = tid & 31;
    const int wave = tid >> 5;
    const int m0 = blockIdx.y * 128;
    const int n0 = blockIdx.x * 128;
    const int mw = (wave >> 1) * 32;   // wave M offset in block tile
    const int nw = (wave & 1) * 64;    // wave N offset in block tile

    v8f acc[2][4];
#pragma unroll
    for (int i = 0; i < 2; ++i)
#pragma unroll
        for (int j = 0; j < 4; ++j)
#pragma unroll
            for (int e = 0; e < 8; ++e) acc[i][j][e] = 0.0f;

    for (int k0 = 0; k0 < DMODEL; k0 += 32) {
        __syncthreads();
        // --- stage A tile: 128 rows x 32 k (fp32 -> bf16) ---
#pragma unroll
        for (int it = 0; it < 4; ++it) {
            int id  = tid + it * 256;          // 0..1023 float4s
            int row = id >> 3;                 // 8 float4 per row
            int c4  = (id & 7) * 4;
            float4 xf = *(const float4*)(X + (size_t)(m0 + row) * DMODEL + k0 + c4);
            unsigned short* p = &lA[row * 48 + c4];
            p[0] = f2bf(xf.x); p[1] = f2bf(xf.y); p[2] = f2bf(xf.z); p[3] = f2bf(xf.w);
        }
        // --- stage B tile: 32 k x 128 n, transposed into [n][k] ---
#pragma unroll
        for (int it = 0; it < 4; ++it) {
            int id = tid + it * 256;
            int kr = id >> 5;                  // 32 float4 per k-row
            int c4 = (id & 31) * 4;
            float4 wf = *(const float4*)(W + (size_t)(k0 + kr) * DMODEL + n0 + c4);
            lB[(c4 + 0) * 48 + kr] = f2bf(wf.x);
            lB[(c4 + 1) * 48 + kr] = f2bf(wf.y);
            lB[(c4 + 2) * 48 + kr] = f2bf(wf.z);
            lB[(c4 + 3) * 48 + kr] = f2bf(wf.w);
        }
        __syncthreads();

        // --- fragments ---
        FragB a[2], b[4];
        const int koffA = (lane < 16) ? 0 : 8;     // A: lanes 0-15 K{0-7,16-23}; 16-31 K{8-15,24-31}
#pragma unroll
        for (int i = 0; i < 2; ++i) {
            const uint4* pr = (const uint4*)&lA[(mw + i * 16 + (lane & 15)) * 48];
            a[i].q[0] = pr[(koffA) >> 3];
            a[i].q[1] = pr[(koffA + 16) >> 3];
        }
        const int koffB = (lane < 16) ? 0 : 8;     // B: lanes 0-15 K{0-7,16-23}; 16-31 K{8-15,24-31}
#pragma unroll
        for (int j = 0; j < 4; ++j) {
            const uint4* pr = (const uint4*)&lB[(nw + j * 16 + (lane & 15)) * 48];
            b[j].q[0] = pr[(koffB >> 3) + 0];
            b[j].q[1] = pr[(koffB >> 3) + 2];
        }
#pragma unroll
        for (int i = 0; i < 2; ++i)
#pragma unroll
            for (int j = 0; j < 4; ++j)
                acc[i][j] = wmma_bf16(a[i], b[j], acc[i][j]);
    }

    // --- epilogue: C layout -> head-split bf16 ---
    const int rowHalf = (lane < 16) ? 0 : 8;
#pragma unroll
    for (int i = 0; i < 2; ++i)
#pragma unroll
        for (int j = 0; j < 4; ++j)
#pragma unroll
            for (int r = 0; r < 8; ++r) {
                int M = m0 + mw + i * 16 + r + rowHalf;
                int N = n0 + nw + j * 16 + (lane & 15);
                int bb = M >> 11, s = M & (S_LEN - 1);
                int h  = N >> 6,  d = N & 63;
                size_t addr;
                if (VTRANS)
                    addr = ((size_t)(bb * HEADS + h) * DHEAD + d) * S_LEN + s;
                else
                    addr = ((size_t)(bb * HEADS + h) * S_LEN + s) * DHEAD + d;
                Out[addr] = f2bf(acc[i][j][r]);
            }
}

// ---------------------------------------------------------------------------
// Flash attention: one wave per 16 query rows, online softmax over 64-key tiles
// Qp,Kp: [B,H,S,64] bf16 ; Vt: [B,H,64,S] bf16 ; Out: [B,S,768] fp32
// ---------------------------------------------------------------------------
__global__ __launch_bounds__(256) void attn_kernel(
    const unsigned short* __restrict__ Qp,
    const unsigned short* __restrict__ Kp,
    const unsigned short* __restrict__ Vt,
    const float* __restrict__ v_mask,
    const float* __restrict__ q_mask,
    float* __restrict__ Out)
{
    __shared__ unsigned short ldsP[8][16 * 72];   // per-wave 16x64 P tile, padded stride
    const int lane = threadIdx.x & 31;
    const int wave = threadIdx.x >> 5;
    const int bh = blockIdx.y;                    // 0..23
    const int b  = bh / HEADS;
    const int h  = bh % HEADS;
    const int q0 = blockIdx.x * 128 + wave * 16;
    const int l15 = lane & 15;

    // Q A-frags (d 0..31 and 32..63), resident all kernel
    FragB qa[2];
    {
        const uint4* qrow = (const uint4*)(Qp + ((size_t)bh * S_LEN + q0 + l15) * DHEAD);
        const int ko = (lane < 16) ? 0 : 1;       // 8-element chunk shift
        qa[0].q[0] = qrow[0 + ko];  qa[0].q[1] = qrow[2 + ko];
        qa[1].q[0] = qrow[4 + ko];  qa[1].q[1] = qrow[6 + ko];
    }

    float m_run[8], l_run[8];
    v8f oacc[4];
#pragma unroll
    for (int r = 0; r < 8; ++r) { m_run[r] = -1e30f; l_run[r] = 0.0f; }
#pragma unroll
    for (int t = 0; t < 4; ++t)
#pragma unroll
        for (int e = 0; e < 8; ++e) oacc[t][e] = 0.0f;

    unsigned short* Prow = &ldsP[wave][0];
    const int half = (lane < 16) ? 0 : 8;

    // loop-invariant row bases
    const unsigned short* kbase = Kp + ((size_t)bh * S_LEN + l15) * DHEAD;   // + (k0+t*16)*DHEAD
    const uint4* vrow[4];
#pragma unroll
    for (int t = 0; t < 4; ++t)
        vrow[t] = (const uint4*)(Vt + ((size_t)bh * DHEAD + t * 16 + l15) * S_LEN);
    const int koB  = (lane < 16) ? 0 : 1;         // K B-frag uint4 shift (d 0-7,16-23 vs 8-15,24-31)
    const int ksO  = (lane < 16) ? 0 : 1;         // V B-frag uint4 shift (keys +0 vs +8)
    const int koA  = (lane < 16) ? 0 : 8;         // A-frag element shift
    const uint4* prP = (const uint4*)(Prow + l15 * 72);

    for (int k0 = 0; k0 < S_LEN; k0 += 64) {
        // prefetch next key tile's K rows into near cache (global_prefetch_b8)
        if (k0 + 64 < S_LEN) {
            __builtin_prefetch(kbase + (size_t)(k0 + 64) * DHEAD, 0, 3);
            __builtin_prefetch(kbase + (size_t)(k0 + 96) * DHEAD, 0, 3);
        }
        // ---- scores S = Q * K^T, 4 column frags of 16 keys ----
        v8f sc[4];
#pragma unroll
        for (int t = 0; t < 4; ++t) {
#pragma unroll
            for (int e = 0; e < 8; ++e) sc[t][e] = 0.0f;
            const uint4* krow = (const uint4*)(kbase + (size_t)(k0 + t * 16) * DHEAD);
            FragB kb;
            kb.q[0] = krow[0 + koB]; kb.q[1] = krow[2 + koB];   // d 0..31
            sc[t] = wmma_bf16(qa[0], kb, sc[t]);
            kb.q[0] = krow[4 + koB]; kb.q[1] = krow[6 + koB];   // d 32..63
            sc[t] = wmma_bf16(qa[1], kb, sc[t]);
        }
        // ---- scale + additive key mask ----
        float addm[4];
#pragma unroll
        for (int t = 0; t < 4; ++t) {
            float vm = v_mask[(size_t)b * S_LEN + k0 + t * 16 + l15];
            addm[t] = -(1.0f - vm) * 1e10f;
        }
        // ---- online softmax (row lives in one VGPR across a 16-lane half) ----
#pragma unroll
        for (int r = 0; r < 8; ++r) {
            float mx = -1e30f;
#pragma unroll
            for (int t = 0; t < 4; ++t) {
                float s = sc[t][r] * 0.125f + addm[t];
                sc[t][r] = s;
                mx = fmaxf(mx, s);
            }
            mx = rowmax16(mx);
            float mnew = fmaxf(m_run[r], mx);
            float c = __expf(m_run[r] - mnew);
            m_run[r] = mnew;
            float rs = 0.0f;
#pragma unroll
            for (int t = 0; t < 4; ++t) {
                float p = __expf(sc[t][r] - mnew);
                sc[t][r] = p;
                rs += p;
            }
            rs = rowsum16(rs);
            l_run[r] = l_run[r] * c + rs;
#pragma unroll
            for (int t = 0; t < 4; ++t) oacc[t][r] *= c;
        }
        // ---- spill P (bf16) to LDS in [row][key] so it can be an A operand ----
#pragma unroll
        for (int t = 0; t < 4; ++t)
#pragma unroll
            for (int r = 0; r < 8; ++r)
                Prow[(r + half) * 72 + t * 16 + l15] = f2bf(sc[t][r]);
        asm volatile("s_wait_dscnt 0" ::: "memory");
        // ---- O += P * V : 2 key chunks x 4 d-frags ----
#pragma unroll
        for (int kc = 0; kc < 2; ++kc) {
            const int cb = kc * 32;
            FragB pa;
            pa.q[0] = prP[(cb + koA) >> 3];
            pa.q[1] = prP[(cb + koA + 16) >> 3];
            const int ks = ((k0 + cb) >> 3) + ksO;
#pragma unroll
            for (int t = 0; t < 4; ++t) {
                FragB vb;
                vb.q[0] = vrow[t][ks]; vb.q[1] = vrow[t][ks + 2];
                oacc[t] = wmma_bf16(pa, vb, oacc[t]);
            }
        }
    }

    // ---- epilogue: normalize, query mask, write fp32 [B,S,768] ----
#pragma unroll
    for (int r = 0; r < 8; ++r) {
        int row = q0 + r + half;
        float qm = q_mask[(size_t)b * S_LEN + row];
        float inv = qm / l_run[r];
#pragma unroll
        for (int t = 0; t < 4; ++t) {
            int d = t * 16 + l15;
            Out[((size_t)b * S_LEN + row) * DMODEL + h * DHEAD + d] = oacc[t][r] * inv;
        }
    }
}

extern "C" void kernel_launch(void* const* d_in, const int* in_sizes, int n_in,
                              void* d_out, int out_size, void* d_ws, size_t ws_size,
                              hipStream_t stream) {
    (void)in_sizes; (void)n_in; (void)out_size; (void)ws_size;
    const float* q      = (const float*)d_in[0];
    const float* k      = (const float*)d_in[1];
    const float* v      = (const float*)d_in[2];
    const float* vmask  = (const float*)d_in[3];
    const float* qmask  = (const float*)d_in[4];
    const float* Wq     = (const float*)d_in[5];
    const float* Wk     = (const float*)d_in[6];
    const float* Wv     = (const float*)d_in[7];
    float* out          = (float*)d_out;

    const size_t perBuf = (size_t)BATCH * HEADS * S_LEN * DHEAD;   // 3.15M elems (bf16)
    unsigned short* Qp = (unsigned short*)d_ws;
    unsigned short* Kp = Qp + perBuf;
    unsigned short* Vt = Kp + perBuf;

    dim3 gp(DMODEL / 128, (BATCH * S_LEN) / 128);   // (6, 32)
    proj_kernel<0><<<gp, 256, 0, stream>>>(q, Wq, Qp);
    proj_kernel<0><<<gp, 256, 0, stream>>>(k, Wk, Kp);
    proj_kernel<1><<<gp, 256, 0, stream>>>(v, Wv, Vt);

    dim3 ga(S_LEN / 128, BATCH * HEADS);            // (16, 24)
    attn_kernel<<<ga, 256, 0, stream>>>(Qp, Kp, Vt, vmask, qmask, out);
}
